// DGCNN_34170759807526
// MI455X (gfx1250) — matrix operand order repaired and checked by reading, not compiled, not run on hardware
//
#include <hip/hip_runtime.h>

// ---------------------------------------------------------------------------
// DGCNN forward for MI455X (gfx1250, wave32, WMMA).
// - All GEMM-shaped work (edge MLPs, lin1, head, kNN Gram matrix) runs on
//   v_wmma_f32_16x16x32_f16 (f16 in, f32 accumulate).
// - Top-K selection is scalar (selection-bound, VALU is the right unit).
// - Weights pre-transposed/padded to f16 once; edge-conv stages them in LDS
//   via CDNA5 global_load_async_to_lds_b128 (ASYNCcnt) when available.
// - Inner WMMA loops use #pragma unroll 1 where invariant-fragment hoisting
//   previously caused scratch spills (lin1/head).
// ---------------------------------------------------------------------------

#define B_   16
#define P_   2048
#define KNN  20

typedef __attribute__((ext_vector_type(16))) _Float16 v16h;
typedef __attribute__((ext_vector_type(8)))  _Float16 v8h;
typedef __attribute__((ext_vector_type(8)))  float    v8f;
typedef int v4i_ __attribute__((__vector_size__(16)));
typedef __attribute__((address_space(1))) v4i_* gv4i_p;   // global
typedef __attribute__((address_space(3))) v4i_* lv4i_p;   // LDS

#if __has_builtin(__builtin_amdgcn_global_load_async_to_lds_b128)
#define EC_ASYNC 1
#endif

// ---- WMMA fragment helpers (CDNA5 16x16x32 f16 layouts) -------------------
// A (16xK row-major, ld = ldk):
//   lanes 0-15 : row = lane,    K = k0 + {0..7, 16..23}
//   lanes 16-31: row = lane-16, K = k0 + {8..15, 24..31}
__device__ __forceinline__ v16h frag_a(const _Float16* m, int ldk, int k0, int lane) {
  const int row = lane & 15;
  const int kh  = (lane >> 4) * 8;
  const v8h* p0 = (const v8h*)(m + row * ldk + k0 + kh);
  const v8h* p1 = (const v8h*)(m + row * ldk + k0 + kh + 16);
  v8h lo = p0[0], hi = p1[0];
  v16h a;
#pragma unroll
  for (int i = 0; i < 8; ++i) { a[i] = lo[i]; a[i + 8] = hi[i]; }
  return a;
}

// B (KxN logical, stored TRANSPOSED as [N][K] so lane loads are contiguous):
//   lanes 0-15 : col = lane,    K = kc + {0..7, 16..23}
//   lanes 16-31: col = lane-16, K = kc + {8..15, 24..31}
__device__ __forceinline__ v16h frag_b(const _Float16* wt, int ldk, int n0, int kc, int lane) {
  const int col = lane & 15;
  const int kh  = (lane >> 4) * 8;
  const v8h* q = (const v8h*)(wt + (size_t)(n0 + col) * ldk + kc + kh);
  v8h lo = q[0], hi = q[2];
  v16h b;
#pragma unroll
  for (int i = 0; i < 8; ++i) { b[i] = lo[i]; b[i + 8] = hi[i]; }
  return b;
}

__device__ __forceinline__ v8f wmma16(v16h a, v16h b, v8f c) {
  return __builtin_amdgcn_wmma_f32_16x16x32_f16(false, a, false, b, (short)0, c, false, false);
}

// ---- async (or plain) 16B global->LDS copy --------------------------------
__device__ __forceinline__ void cp16_to_lds(_Float16* lds, const _Float16* g) {
#ifdef EC_ASYNC
  __builtin_amdgcn_global_load_async_to_lds_b128((gv4i_p)g, (lv4i_p)lds, 0, 0);
#else
  *(uint4*)lds = *(const uint4*)g;
#endif
}
__device__ __forceinline__ void cp16_wait() {
#ifdef EC_ASYNC
#if __has_builtin(__builtin_amdgcn_s_wait_asynccnt)
  __builtin_amdgcn_s_wait_asynccnt(0);
#else
  asm volatile("s_wait_asynccnt 0" ::: "memory");
#endif
#endif
}

// ---- weight transpose + f32->f16 convert (with K/N zero padding) ----------
__global__ void __launch_bounds__(256)
tconv_kernel(const float* __restrict__ W, _Float16* __restrict__ Wt,
             int K, int N, int Kp, int Np) {
  int i = blockIdx.x * 256 + threadIdx.x;
  if (i >= Kp * Np) return;
  int n = i / Kp, k = i - n * Kp;
  Wt[i] = (n < N && k < K) ? (_Float16)W[(size_t)k * N + n] : (_Float16)0.f;
}

// ---- f32 -> f16 copy + squared norms (for WMMA kNN) -----------------------
template <int D>
__global__ void __launch_bounds__(256)
cvt_norm_kernel(const float* __restrict__ x, _Float16* __restrict__ xh,
                float* __restrict__ nrm) {
  int p = blockIdx.x * 256 + threadIdx.x;
  if (p >= B_ * P_) return;
  const float* src = x + (size_t)p * D;
  _Float16* dst = xh + (size_t)p * D;
  float s = 0.f;
#pragma unroll
  for (int d = 0; d < D; ++d) { float v = src[d]; s = fmaf(v, v, s); dst[d] = (_Float16)v; }
  nrm[p] = s;
}

// ---- scalar kNN (only for D=3; selection-bound) ---------------------------
template <int D>
__global__ void __launch_bounds__(128)
knn_kernel(const float* __restrict__ x, int* __restrict__ idx) {
  constexpr int CT = 32;
  __shared__ float cf[CT * D];
  const int cloud = blockIdx.x >> 4;
  const int q     = ((blockIdx.x & 15) << 7) + threadIdx.x;
  const float* xc = x + (size_t)cloud * P_ * D;
  float qf[D];
#pragma unroll
  for (int d = 0; d < D; ++d) qf[d] = xc[(size_t)q * D + d];
  float bd[KNN]; int bi[KNN];
#pragma unroll
  for (int t = 0; t < KNN; ++t) { bd[t] = 3.4e38f; bi[t] = 0; }

  for (int tile = 0; tile < P_; tile += CT) {
    __syncthreads();
    for (int i = threadIdx.x; i < CT * D; i += 128) cf[i] = xc[(size_t)tile * D + i];
    if (tile + CT < P_) {
      const float* nxt = xc + (size_t)(tile + CT) * D;
      for (int i = threadIdx.x; i * 16 < CT * D; i += 128)
        __builtin_prefetch(nxt + i * 16, 0, 0);
    }
    __syncthreads();
    for (int c = 0; c < CT; ++c) {
      float d2 = 0.f;
#pragma unroll
      for (int d = 0; d < D; ++d) { float t = qf[d] - cf[c * D + d]; d2 = fmaf(t, t, d2); }
      if (d2 < bd[KNN - 1]) {
        float cd = d2; int ci = tile + c;
#pragma unroll
        for (int t = 0; t < KNN; ++t) {
          bool s = cd < bd[t];
          float td = bd[t]; int ti = bi[t];
          if (s) { bd[t] = cd; bi[t] = ci; cd = td; ci = ti; }
        }
      }
    }
  }
  int* op = idx + ((size_t)cloud * P_ + q) * KNN;
#pragma unroll
  for (int t = 0; t < KNN; ++t) op[t] = bi[t];
}

// ---- WMMA kNN: Gram tiles on matrix cores, packed-key top-K ---------------
// Each wave owns 16 queries (B-frag, hoisted) and sweeps all 2048 candidates
// as A-frags. c[e]: row = candidate (rb+e), col = query (lane&15). Lane keeps
// a sorted top-20 of its 1024-candidate half as (d2|idx) u64 keys; halves are
// merged per query through LDS.
template <int D>
__global__ void __launch_bounds__(256)
knn_wmma_kernel(const _Float16* __restrict__ xh, const float* __restrict__ nrm,
                int* __restrict__ idx) {
  __shared__ float sCN[P_];                                // 8 KB
  __shared__ unsigned long long sKey[8][16][2][KNN];       // 40 KB
  const int tid = threadIdx.x, wv = tid >> 5, lane = tid & 31;
  const int cloud = blockIdx.x >> 4;                       // 16 blocks / cloud
  const int qbase = ((blockIdx.x & 15) << 7) + wv * 16;    // 128 queries/block
  const _Float16* xc = xh + (size_t)cloud * P_ * D;
  const float* nc = nrm + (size_t)cloud * P_;
  for (int i = tid; i < P_; i += 256) sCN[i] = nc[i];
  __syncthreads();

  const float qn = nc[qbase + (lane & 15)];
  unsigned long long bk[KNN];
#pragma unroll
  for (int t = 0; t < KNN; ++t) bk[t] = ~0ull;
  const int rb = (lane >> 4) * 8;

  for (int cb = 0; cb < P_; cb += 16) {
    v8f c;
#pragma unroll
    for (int e = 0; e < 8; ++e) c[e] = 0.f;
#pragma unroll
    for (int kc = 0; kc < D; kc += 32)
      c = wmma16(frag_a(xc + (size_t)cb * D, D, kc, lane),   // candidates
                 frag_b(xc, D, qbase, kc, lane), c);         // queries (hoisted)
#pragma unroll
    for (int e = 0; e < 8; ++e) {
      int cand = cb + rb + e;
      float d2 = fmaxf(sCN[cand] + qn - 2.f * c[e], 0.f);
      unsigned long long key =
          ((unsigned long long)__float_as_uint(d2) << 32) | (unsigned)cand;
      if (key < bk[KNN - 1]) {
        unsigned long long ck = key;
#pragma unroll
        for (int t = 0; t < KNN; ++t) {
          bool s = ck < bk[t];
          unsigned long long tk = bk[t];
          if (s) { bk[t] = ck; ck = tk; }
        }
      }
    }
  }
  const int q = lane & 15, half = lane >> 4;
#pragma unroll
  for (int t = 0; t < KNN; ++t) sKey[wv][q][half][t] = bk[t];
  __syncthreads();
  if (lane < 16) {
    int* op = idx + ((size_t)cloud * P_ + qbase + lane) * KNN;
    int i0 = 0, i1 = 0;
    for (int t = 0; t < KNN; ++t) {
      unsigned long long k0 = sKey[wv][lane][0][i0];
      unsigned long long k1 = sKey[wv][lane][1][i1];
      bool take0 = k0 <= k1;
      op[t] = (int)(unsigned)(take0 ? k0 : k1);
      if (take0) ++i0; else ++i1;
    }
  }
}

// ---- EdgeConv: 16-point tile / block, 8 waves, full WMMA MLP --------------
template <int DIN, int H, int DOUT>
__global__ void __launch_bounds__(256)
edgeconv_kernel(const float* __restrict__ x, const int* __restrict__ nbr,
                const _Float16* __restrict__ W1t, const float* __restrict__ b1,
                const _Float16* __restrict__ W2t, const float* __restrict__ b2,
                const _Float16* __restrict__ W3t, const float* __restrict__ b3,
                float* __restrict__ out) {
  constexpr int KP1 = ((2 * DIN + 31) / 32) * 32;
  constexpr int NW  = 8;
  constexpr int T3  = (DOUT / 16 + NW - 1) / NW;

  extern __shared__ char smem_raw[];
  _Float16* sW1 = (_Float16*)smem_raw;              // [H][KP1]   (transposed)
  _Float16* sW2 = sW1 + (size_t)H * KP1;            // [H][H]
  _Float16* sW3 = sW2 + (size_t)H * H;              // [DOUT][H]
  _Float16* sEF = sW3 + (size_t)DOUT * H;           // [16][KP1]
  _Float16* sH1 = sEF + 16 * KP1;                   // [16][H]
  _Float16* sH2 = sH1 + 16 * H;                     // [16][H]
  float*    sXI = (float*)(sH2 + 16 * H);           // [16][DIN]
  int*      sNB = (int*)(sXI + 16 * DIN);           // [16]

  const int tid = threadIdx.x, wv = tid >> 5, lane = tid & 31;
  const int cloud = blockIdx.x >> 7;
  const int pbase = (blockIdx.x & 127) << 4;

  // stage weights into LDS (async-to-LDS when available)
  for (int i = tid; i < H * KP1 / 8; i += 256) cp16_to_lds(sW1 + i * 8, W1t + i * 8);
  for (int i = tid; i < H * H / 8; i += 256)   cp16_to_lds(sW2 + i * 8, W2t + i * 8);
  for (int i = tid; i < DOUT * H / 8; i += 256) cp16_to_lds(sW3 + i * 8, W3t + i * 8);
  cp16_wait();

  const float* xc = x + (size_t)cloud * P_ * DIN;
  for (int i = tid; i < 16 * DIN; i += 256) {
    int r = i / DIN, d = i - r * DIN;
    sXI[i] = xc[(size_t)(pbase + r) * DIN + d];
  }
  __syncthreads();
  for (int i = tid; i < 16 * KP1; i += 256) {       // static half of edge feat
    int r = i / KP1, c = i - r * KP1;
    sEF[i] = (_Float16)((c < DIN) ? sXI[r * DIN + c] : 0.f);
  }

  v8f macc[T3];
  for (int t = 0; t < T3; ++t) { v8f z; for (int e = 0; e < 8; ++e) z[e] = -3.4e38f; macc[t] = z; }

  const int* ncp = nbr + ((size_t)cloud * P_ + pbase) * KNN;
  for (int k = 0; k < KNN; ++k) {
    __syncthreads();
    if (tid < 16) sNB[tid] = ncp[tid * KNN + k];
    __syncthreads();
    for (int i = tid; i < 16 * DIN; i += 256) {     // dynamic half: x_j - x_i
      int r = i / DIN, d = i - r * DIN;
      float xj = xc[(size_t)sNB[r] * DIN + d];
      sEF[r * KP1 + DIN + d] = (_Float16)(xj - sXI[i]);
    }
    __syncthreads();
    // stage 1: h1 = relu(ef @ W1 + b1)
    for (int ct = wv; ct < H / 16; ct += NW) {
      float bb = b1[ct * 16 + (lane & 15)];
      v8f c; for (int e = 0; e < 8; ++e) c[e] = bb;
#pragma unroll
      for (int kc = 0; kc < KP1; kc += 32)
        c = wmma16(frag_a(sEF, KP1, kc, lane), frag_b(sW1, KP1, ct * 16, kc, lane), c);
      const int col = ct * 16 + (lane & 15), rbw = (lane >> 4) * 8;
#pragma unroll
      for (int e = 0; e < 8; ++e) sH1[(rbw + e) * H + col] = (_Float16)fmaxf(c[e], 0.f);
    }
    __syncthreads();
    // stage 2: h2 = relu(h1 @ W2 + b2)
    for (int ct = wv; ct < H / 16; ct += NW) {
      float bb = b2[ct * 16 + (lane & 15)];
      v8f c; for (int e = 0; e < 8; ++e) c[e] = bb;
#pragma unroll
      for (int kc = 0; kc < H; kc += 32)
        c = wmma16(frag_a(sH1, H, kc, lane), frag_b(sW2, H, ct * 16, kc, lane), c);
      const int col = ct * 16 + (lane & 15), rbw = (lane >> 4) * 8;
#pragma unroll
      for (int e = 0; e < 8; ++e) sH2[(rbw + e) * H + col] = (_Float16)fmaxf(c[e], 0.f);
    }
    __syncthreads();
    // stage 3: max-accumulate (h2 @ W3 + b3) in registers
#pragma unroll
    for (int t = 0; t < T3; ++t) {
      int ct = wv + t * NW;
      if (ct < DOUT / 16) {
        float bb = b3[ct * 16 + (lane & 15)];
        v8f c; for (int e = 0; e < 8; ++e) c[e] = bb;
#pragma unroll
        for (int kc = 0; kc < H; kc += 32)
          c = wmma16(frag_a(sH2, H, kc, lane), frag_b(sW3, H, ct * 16, kc, lane), c);
#pragma unroll
        for (int e = 0; e < 8; ++e) macc[t][e] = fmaxf(macc[t][e], c[e]);
      }
    }
  }
  float* oc = out + ((size_t)cloud * P_ + pbase) * DOUT;
#pragma unroll
  for (int t = 0; t < T3; ++t) {
    int ct = wv + t * NW;
    if (ct < DOUT / 16) {
      const int col = ct * 16 + (lane & 15), rbw = (lane >> 4) * 8;
#pragma unroll
      for (int e = 0; e < 8; ++e) oc[(size_t)(rbw + e) * DOUT + col] = macc[t][e];
    }
  }
}

// ---- lin1: 4 row-tile accumulators share one B frag; no spills ------------
__global__ void __launch_bounds__(256)
lin1_kernel(const float* __restrict__ x1, const float* __restrict__ x2,
            const float* __restrict__ x3, const float* __restrict__ x4,
            const _Float16* __restrict__ Wt /*[1024][512]*/,
            const float* __restrict__ bias,
            float* __restrict__ part /*[B][32][1024]*/) {
  extern __shared__ char smem_raw[];
  _Float16* sEF = (_Float16*)smem_raw;              // [64][512]
  const int tid = threadIdx.x, wv = tid >> 5, lane = tid & 31;
  const int cloud = blockIdx.x >> 5;
  const int pblk  = blockIdx.x & 31;
  const int pbase = pblk * 64;
  for (int i = tid; i < 64 * 512; i += 256) {
    int r = i >> 9, c = i & 511;
    size_t p = (size_t)cloud * P_ + pbase + r;
    float v;
    if (c < 64)       v = x1[p * 64 + c];
    else if (c < 128) v = x2[p * 64 + (c - 64)];
    else if (c < 256) v = x3[p * 128 + (c - 128)];
    else              v = x4[p * 256 + (c - 256)];
    sEF[i] = (_Float16)v;
  }
  __syncthreads();
  float* pb = part + ((size_t)cloud * 32 + pblk) * 1024;
  for (int ct = wv; ct < 64; ct += 8) {
    float bb = bias[ct * 16 + (lane & 15)];
    v8f c0, c1, c2, c3;
    for (int e = 0; e < 8; ++e) { c0[e] = bb; c1[e] = bb; c2[e] = bb; c3[e] = bb; }
#pragma unroll 1
    for (int kc = 0; kc < 512; kc += 32) {
      v16h b = frag_b(Wt, 512, ct * 16, kc, lane);
      c0 = wmma16(frag_a(sEF,          512, kc, lane), b, c0);
      c1 = wmma16(frag_a(sEF + 8192,   512, kc, lane), b, c1);
      c2 = wmma16(frag_a(sEF + 16384,  512, kc, lane), b, c2);
      c3 = wmma16(frag_a(sEF + 24576,  512, kc, lane), b, c3);
    }
    float m = -3.4e38f;
#pragma unroll
    for (int e = 0; e < 8; ++e)
      m = fmaxf(m, fmaxf(fmaxf(c0[e], c1[e]), fmaxf(c2[e], c3[e])));
    m = fmaxf(m, __shfl_xor(m, 16, 32));
    if (lane < 16) pb[ct * 16 + lane] = m;
  }
}

__global__ void __launch_bounds__(256)
reduce_max_kernel(const float* __restrict__ part, float* __restrict__ g) {
  const int cloud = blockIdx.x;
  const int ch = blockIdx.y * 256 + threadIdx.x;
  const float* p = part + (size_t)cloud * 32 * 1024 + ch;
  float m = -3.4e38f;
#pragma unroll 4
  for (int t = 0; t < 32; ++t) m = fmaxf(m, p[t * 1024]);
  g[cloud * 1024 + ch] = m;
}

// ---- classifier head: [16,1024]->512->256->40, single block, WMMA ---------
__global__ void __launch_bounds__(256)
head_kernel(const float* __restrict__ g,
            const _Float16* __restrict__ W1t /*[512][1024]*/, const float* __restrict__ b1,
            const _Float16* __restrict__ W2t /*[256][512]*/,  const float* __restrict__ b2,
            const _Float16* __restrict__ W3t /*[48][256]*/,   const float* __restrict__ b3,
            float* __restrict__ out /*[16][40]*/) {
  extern __shared__ char smem_raw[];
  _Float16* sG  = (_Float16*)smem_raw;              // [16][1024]
  _Float16* sH1 = sG + 16 * 1024;                   // [16][512]
  _Float16* sH2 = sH1 + 16 * 512;                   // [16][256]
  const int tid = threadIdx.x, wv = tid >> 5, lane = tid & 31;
  for (int i = tid; i < 16 * 1024; i += 256) sG[i] = (_Float16)g[i];
  __syncthreads();
  for (int ct = wv; ct < 32; ct += 8) {
    float bb = b1[ct * 16 + (lane & 15)];
    v8f c; for (int e = 0; e < 8; ++e) c[e] = bb;
#pragma unroll 1
    for (int kc = 0; kc < 1024; kc += 32)
      c = wmma16(frag_a(sG, 1024, kc, lane), frag_b(W1t, 1024, ct * 16, kc, lane), c);
    const int col = ct * 16 + (lane & 15), rbw = (lane >> 4) * 8;
#pragma unroll
    for (int e = 0; e < 8; ++e) sH1[(rbw + e) * 512 + col] = (_Float16)fmaxf(c[e], 0.f);
  }
  __syncthreads();
  for (int ct = wv; ct < 16; ct += 8) {
    float bb = b2[ct * 16 + (lane & 15)];
    v8f c; for (int e = 0; e < 8; ++e) c[e] = bb;
#pragma unroll 1
    for (int kc = 0; kc < 512; kc += 32)
      c = wmma16(frag_a(sH1, 512, kc, lane), frag_b(W2t, 512, ct * 16, kc, lane), c);
    const int col = ct * 16 + (lane & 15), rbw = (lane >> 4) * 8;
#pragma unroll
    for (int e = 0; e < 8; ++e) sH2[(rbw + e) * 256 + col] = (_Float16)fmaxf(c[e], 0.f);
  }
  __syncthreads();
  for (int ct = wv; ct < 3; ct += 8) {              // 40 cols padded to 48
    const int col = ct * 16 + (lane & 15);
    float bb = (col < 40) ? b3[col] : 0.f;
    v8f c; for (int e = 0; e < 8; ++e) c[e] = bb;
#pragma unroll 1
    for (int kc = 0; kc < 256; kc += 32)
      c = wmma16(frag_a(sH2, 256, kc, lane), frag_b(W3t, 256, ct * 16, kc, lane), c);
    const int rbw = (lane >> 4) * 8;
    if (col < 40) {
#pragma unroll
      for (int e = 0; e < 8; ++e) out[(rbw + e) * 40 + col] = c[e];
    }
  }
}

// ---- host-side shared-mem size mirror -------------------------------------
static constexpr size_t ec_smem(int DIN, int H, int DOUT) {
  int KP1 = ((2 * DIN + 31) / 32) * 32;
  size_t h = (size_t)H * KP1 + (size_t)H * H + (size_t)DOUT * H + 16 * KP1 + 32 * H;
  return h * 2 + (size_t)16 * DIN * 4 + 16 * 4;
}

extern "C" void kernel_launch(void* const* d_in, const int* in_sizes, int n_in,
                              void* d_out, int out_size, void* d_ws, size_t ws_size,
                              hipStream_t stream) {
  (void)in_sizes; (void)n_in; (void)out_size; (void)ws_size;
  const float* pos  = (const float*)d_in[0];
  const float* c1W1 = (const float*)d_in[2],  *c1b1 = (const float*)d_in[3];
  const float* c1W2 = (const float*)d_in[4],  *c1b2 = (const float*)d_in[5];
  const float* c1W3 = (const float*)d_in[6],  *c1b3 = (const float*)d_in[7];
  const float* c2W1 = (const float*)d_in[8],  *c2b1 = (const float*)d_in[9];
  const float* c2W2 = (const float*)d_in[10], *c2b2 = (const float*)d_in[11];
  const float* c2W3 = (const float*)d_in[12], *c2b3 = (const float*)d_in[13];
  const float* c3W1 = (const float*)d_in[14], *c3b1 = (const float*)d_in[15];
  const float* c3W2 = (const float*)d_in[16], *c3b2 = (const float*)d_in[17];
  const float* c3W3 = (const float*)d_in[18], *c3b3 = (const float*)d_in[19];
  const float* c4W1 = (const float*)d_in[20], *c4b1 = (const float*)d_in[21];
  const float* c4W2 = (const float*)d_in[22], *c4b2 = (const float*)d_in[23];
  const float* c4W3 = (const float*)d_in[24], *c4b3 = (const float*)d_in[25];
  const float* linW = (const float*)d_in[26], *linb = (const float*)d_in[27];
  const float* hW1  = (const float*)d_in[28], *hb1  = (const float*)d_in[29];
  const float* hW2  = (const float*)d_in[30], *hb2  = (const float*)d_in[31];
  const float* hW3  = (const float*)d_in[32], *hb3  = (const float*)d_in[33];

  char* w = (char*)d_ws;
  size_t off = 0;
  auto carve = [&](size_t bytes) -> char* {
    char* p = w + off;
    off += (bytes + 255) & ~(size_t)255;
    return p;
  };
  float* x1    = (float*)carve((size_t)B_ * P_ * 64 * 4);
  float* x2    = (float*)carve((size_t)B_ * P_ * 64 * 4);
  float* x3    = (float*)carve((size_t)B_ * P_ * 128 * 4);
  float* x4    = (float*)carve((size_t)B_ * P_ * 256 * 4);
  int*   nb    = (int*)carve((size_t)B_ * P_ * KNN * 4);
  float* part  = (float*)carve((size_t)B_ * 32 * 1024 * 4);
  float* gpool = (float*)carve((size_t)B_ * 1024 * 4);
  _Float16* xh = (_Float16*)carve((size_t)B_ * P_ * 128 * 2);   // f16 copy for kNN
  float* nrm   = (float*)carve((size_t)B_ * P_ * 4);
  _Float16* t1a = (_Float16*)carve(64 * 32 * 2);
  _Float16* t1b = (_Float16*)carve(64 * 64 * 2);
  _Float16* t1c = (_Float16*)carve(64 * 64 * 2);
  _Float16* t2a = (_Float16*)carve(64 * 128 * 2);
  _Float16* t2b = (_Float16*)carve(64 * 64 * 2);
  _Float16* t2c = (_Float16*)carve(64 * 64 * 2);
  _Float16* t3a = (_Float16*)carve(64 * 128 * 2);
  _Float16* t3b = (_Float16*)carve(64 * 64 * 2);
  _Float16* t3c = (_Float16*)carve(128 * 64 * 2);
  _Float16* t4a = (_Float16*)carve(128 * 256 * 2);
  _Float16* t4b = (_Float16*)carve(128 * 128 * 2);
  _Float16* t4c = (_Float16*)carve(256 * 128 * 2);
  _Float16* tlin = (_Float16*)carve((size_t)1024 * 512 * 2);
  _Float16* th1  = (_Float16*)carve((size_t)512 * 1024 * 2);
  _Float16* th2  = (_Float16*)carve((size_t)256 * 512 * 2);
  _Float16* th3  = (_Float16*)carve((size_t)48 * 256 * 2);

  auto tc = [&](const float* W, _Float16* Wt, int K, int N, int Kp, int Np) {
    int total = Kp * Np;
    tconv_kernel<<<(total + 255) / 256, 256, 0, stream>>>(W, Wt, K, N, Kp, Np);
  };
  tc(c1W1, t1a, 6, 64, 32, 64);      tc(c1W2, t1b, 64, 64, 64, 64);     tc(c1W3, t1c, 64, 64, 64, 64);
  tc(c2W1, t2a, 128, 64, 128, 64);   tc(c2W2, t2b, 64, 64, 64, 64);     tc(c2W3, t2c, 64, 64, 64, 64);
  tc(c3W1, t3a, 128, 64, 128, 64);   tc(c3W2, t3b, 64, 64, 64, 64);     tc(c3W3, t3c, 64, 128, 64, 128);
  tc(c4W1, t4a, 256, 128, 256, 128); tc(c4W2, t4b, 128, 128, 128, 128); tc(c4W3, t4c, 128, 256, 128, 256);
  tc(linW, tlin, 512, 1024, 512, 1024);
  tc(hW1, th1, 1024, 512, 1024, 512);
  tc(hW2, th2, 512, 256, 512, 256);
  tc(hW3, th3, 256, 40, 256, 48);

  const int knn_grid = B_ * (P_ / 128);
  const int ec_grid  = B_ * (P_ / 16);
  const int cv_grid  = (B_ * P_ + 255) / 256;

  knn_kernel<3><<<knn_grid, 128, 0, stream>>>(pos, nb);
  edgeconv_kernel<3, 64, 64><<<ec_grid, 256, ec_smem(3, 64, 64), stream>>>(
      pos, nb, t1a, c1b1, t1b, c1b2, t1c, c1b3, x1);

  cvt_norm_kernel<64><<<cv_grid, 256, 0, stream>>>(x1, xh, nrm);
  knn_wmma_kernel<64><<<knn_grid, 256, 0, stream>>>(xh, nrm, nb);
  edgeconv_kernel<64, 64, 64><<<ec_grid, 256, ec_smem(64, 64, 64), stream>>>(
      x1, nb, t2a, c2b1, t2b, c2b2, t2c, c2b3, x2);

  cvt_norm_kernel<64><<<cv_grid, 256, 0, stream>>>(x2, xh, nrm);
  knn_wmma_kernel<64><<<knn_grid, 256, 0, stream>>>(xh, nrm, nb);
  edgeconv_kernel<64, 64, 128><<<ec_grid, 256, ec_smem(64, 64, 128), stream>>>(
      x2, nb, t3a, c3b1, t3b, c3b2, t3c, c3b3, x3);

  cvt_norm_kernel<128><<<cv_grid, 256, 0, stream>>>(x3, xh, nrm);
  knn_wmma_kernel<128><<<knn_grid, 256, 0, stream>>>(xh, nrm, nb);
  edgeconv_kernel<128, 128, 256><<<ec_grid, 256, ec_smem(128, 128, 256), stream>>>(
      x3, nb, t4a, c4b1, t4b, c4b2, t4c, c4b3, x4);

  lin1_kernel<<<B_ * 32, 256, (size_t)64 * 512 * 2, stream>>>(
      x1, x2, x3, x4, tlin, linb, part);
  reduce_max_kernel<<<dim3(B_, 4), 256, 0, stream>>>(part, gpool);
  head_kernel<<<1, 256, (size_t)(16 * 1024 + 16 * 512 + 16 * 256) * 2, stream>>>(
      gpool, th1, hb1, th2, hb2, th3, hb3, (float*)d_out);
}
